// AttentionLayer_65274912965281
// MI455X (gfx1250) — compile-verified, matrix operand order checked by reading, not run on hardware
//
#include <hip/hip_runtime.h>

// ---------------------------------------------------------------------------
// Types for CDNA5 WMMA / TDM (gfx1250, wave32)
// ---------------------------------------------------------------------------
typedef __attribute__((ext_vector_type(16))) __bf16 v16bf;
typedef __attribute__((ext_vector_type(8)))  __bf16 v8bf;
typedef __attribute__((ext_vector_type(8)))  float  v8f;
typedef __attribute__((ext_vector_type(4)))  unsigned int u32x4;
typedef __attribute__((ext_vector_type(8)))  int i32x8;
typedef __attribute__((ext_vector_type(4)))  int i32x4;

#define TEXT_DIM 768
#define IMAGE_DIM 1024
#define HIDDEN 2048
#define FFD 128
#define HEADS 8
#define HEAD_DIM 256
#define BATCH 16
#define LT 512
#define LI 576

// float -> bf16 round-to-nearest-even
static __device__ __forceinline__ __bf16 f2bf(float f) {
  unsigned u = __builtin_bit_cast(unsigned, f);
  u += 0x7FFFu + ((u >> 16) & 1u);
  unsigned short h = (unsigned short)(u >> 16);
  return __builtin_bit_cast(__bf16, h);
}

// Load a 16-element bf16 WMMA fragment for this lane.
// CDNA5 ISA 16-bit A/B fragment striping (16x16x32):
//   lanes 0-15 : K = {0..7, 16..23},  lanes 16-31 : K = {8..15, 24..31}
// Caller passes p = rowBase + k0 + kbase (kbase = (lane>=16)*8): two
// contiguous 16-byte chunks at p and p+16 elements.
static __device__ __forceinline__ v16bf ldfrag(const __bf16* p) {
  v8bf lo = *(const v8bf*)(p);
  v8bf hi = *(const v8bf*)(p + 16);
  v16bf r;
#pragma unroll
  for (int i = 0; i < 8; ++i) { r[i] = lo[i]; r[i + 8] = hi[i]; }
  return r;
}

// ---------------------------------------------------------------------------
// Elementwise converts
// ---------------------------------------------------------------------------
__global__ void cvt_f32_bf16(const float* __restrict__ in,
                             __bf16* __restrict__ out, int n) {
  int i = blockIdx.x * blockDim.x + threadIdx.x;
  if (i < n) out[i] = f2bf(in[i]);
}

// W is [Kdim, Ndim] row-major fp32; Wt is [Ndim, Kdim] row-major bf16.
__global__ void transpose_cvt(const float* __restrict__ W,
                              __bf16* __restrict__ Wt, int Kdim, int Ndim) {
  int i = blockIdx.x * blockDim.x + threadIdx.x;
  if (i >= Kdim * Ndim) return;
  int n = i / Kdim;
  int k = i - n * Kdim;
  Wt[i] = f2bf(W[(size_t)k * Ndim + n]);
}

// ---------------------------------------------------------------------------
// Issue a TDM load of a 64-row x 32-col bf16 tile (rows n0.., cols kt*32..)
// from row-major Bt[N,K] into LDS at lds_addr. 2D descriptor, D# groups per
// CDNA5 ISA. Tracked by TENSORcnt.
// ---------------------------------------------------------------------------
static __device__ __forceinline__ void tdm_load_tile(
    const __bf16* Bt, int K, int N, int n0, int kt, unsigned lds_addr) {
  unsigned long long ga =
      (unsigned long long)(const void*)(Bt + (size_t)n0 * K + (size_t)kt * 32);
  u32x4 g0;
  g0[0] = 1u;                                   // count=1 valid descriptor
  g0[1] = lds_addr;                             // lds_addr (bytes)
  g0[2] = (unsigned)(ga & 0xFFFFFFFFull);       // global_addr[31:0]
  g0[3] = (unsigned)(ga >> 32) | (2u << 30);    // global_addr[56:32] | type=2

  unsigned td0 = (unsigned)K;                   // tensor dim0 (elements)
  unsigned td1 = (unsigned)N;                   // tensor dim1 (rows)
  unsigned long long s0 = (unsigned long long)K;  // dim0 stride (elements)
  i32x8 g1;
  g1[0] = (int)(1u << 16);                      // wg_mask=0, data_size=1 (2B)
  g1[1] = (int)((td0 & 0xFFFFu) << 16);         // tensor_dim0[15:0]
  g1[2] = (int)(((td0 >> 16) & 0xFFFFu) | ((td1 & 0xFFFFu) << 16));
  g1[3] = (int)(((td1 >> 16) & 0xFFFFu) | (32u << 16));   // tile_dim0=32
  g1[4] = (int)64u;                             // tile_dim1=64, tile_dim2=0
  g1[5] = (int)(unsigned)(s0 & 0xFFFFFFFFull);  // dim0 stride [31:0]
  g1[6] = (int)(unsigned)((s0 >> 32) & 0xFFFFull);  // stride[47:32], s1=0
  g1[7] = 0;
  i32x4 g2 = {0, 0, 0, 0};
  i32x4 g3 = {0, 0, 0, 0};
#if __clang_major__ >= 23
  i32x8 g4 = {0, 0, 0, 0, 0, 0, 0, 0};
  __builtin_amdgcn_tensor_load_to_lds(g0, g1, g2, g3, g4, 0);
#else
  __builtin_amdgcn_tensor_load_to_lds(g0, g1, g2, g3, 0);
#endif
}

// ---------------------------------------------------------------------------
// WMMA GEMM with TDM-staged, double-buffered B tiles in LDS.
//   C[M,N] = A[M,K] @ Bt[N,K]^T + bias   (A,Bt bf16 row-major)
// Block = 4 waves -> 64x64 C tile; B tile (64x32) shared via LDS.
// vmode==0: store Cf(f32)/Cb(bf16) with optional relu / f32 residual.
// vmode==1: scatter-store bf16 into per-head transposed V layout via Cb:
//           Vt[b][h][d][key] (key contiguous).
// ---------------------------------------------------------------------------
__global__ __launch_bounds__(128) void gemm_wmma_tdm(
    const __bf16* __restrict__ A, const __bf16* __restrict__ Bt,
    const float* __restrict__ bias,
    float* __restrict__ Cf, __bf16* __restrict__ Cb,
    const float* __restrict__ Res,
    int M, int N, int K, int relu, int vmode) {
  __shared__ __align__(16) __bf16 Btile[2][64][32];

  const int lane = threadIdx.x & 31;
  const int wave = threadIdx.x >> 5;
  const int m0 = blockIdx.y * 64 + wave * 16;
  const int n0 = blockIdx.x * 64;
  const int half = lane >> 4;
  const int ln   = lane & 15;
  const int kb   = half * 8;
  const int nk   = K >> 5;

  const unsigned lds0 =
      (unsigned)(unsigned long long)(const void*)&Btile[0][0][0];
  const unsigned lds1 =
      (unsigned)(unsigned long long)(const void*)&Btile[1][0][0];

  if (wave == 0) tdm_load_tile(Bt, K, N, n0, 0, lds0);

  const __bf16* arow = A + (size_t)(m0 + ln) * K;
  v8f acc[4] = {v8f{}, v8f{}, v8f{}, v8f{}};

  for (int i = 0; i < nk; ++i) {
    __syncthreads();  // previous compute done -> safe to overwrite other buf
    if (wave == 0) {
      if (i + 1 < nk) {
        tdm_load_tile(Bt, K, N, n0, i + 1, ((i + 1) & 1) ? lds1 : lds0);
        __builtin_amdgcn_s_wait_tensorcnt(1);  // older DMA (tile i) complete
      } else {
        __builtin_amdgcn_s_wait_tensorcnt(0);
      }
    }
    __syncthreads();  // tile i visible to all waves

    v16bf af = ldfrag(arow + i * 32 + kb);
    const __bf16* btile = &Btile[i & 1][0][0];
#pragma unroll
    for (int t = 0; t < 4; ++t) {
      v16bf bf = ldfrag(btile + (t * 16 + ln) * 32 + kb);
      acc[t] = __builtin_amdgcn_wmma_f32_16x16x32_bf16(
          false, af, false, bf, (short)0, acc[t], false, false);
    }
  }

  if (vmode == 0) {
#pragma unroll
    for (int t = 0; t < 4; ++t) {
#pragma unroll
      for (int v = 0; v < 8; ++v) {
        int m = m0 + v + 8 * half;  // C layout: lanes 0-15 M=v, 16-31 M=v+8
        int n = n0 + t * 16 + ln;
        float c = acc[t][v] + bias[n];
        if (relu) c = fmaxf(c, 0.0f);
        if (Res) c += Res[(size_t)m * N + n];
        if (Cf) Cf[(size_t)m * N + n] = c;
        if (Cb) Cb[(size_t)m * N + n] = f2bf(c);
      }
    }
  } else {
#pragma unroll
    for (int t = 0; t < 4; ++t) {
#pragma unroll
      for (int v = 0; v < 8; ++v) {
        int m = m0 + v + 8 * half;   // image row: b*LI + key
        int n = n0 + t * 16 + ln;    // hidden index: h*256 + d
        int b = m / LI, key = m - b * LI;
        int h = n >> 8, d = n & 255;
        float c = acc[t][v] + bias[n];
        Cb[(((size_t)b * HEADS + h) * HEAD_DIM + d) * LI + key] = f2bf(c);
      }
    }
  }
}

// ---------------------------------------------------------------------------
// Attention: one wave per (16 query rows, head, batch).
//   S = (Q @ K^T) / 16  -> LDS fp32 [16][576]
//   softmax rows (wave-parallel, shfl reductions) -> P bf16 in LDS
//   O = P @ V (Vt layout) -> X[b, q, h*256+d] bf16
// ---------------------------------------------------------------------------
__global__ __launch_bounds__(32) void attn_kernel(
    const __bf16* __restrict__ Q, const __bf16* __restrict__ K,
    const __bf16* __restrict__ Vt, __bf16* __restrict__ X) {
  __shared__ __align__(16) float  S[16 * LI];
  __shared__ __align__(16) __bf16 P[16 * LI];
  __shared__ float rinv[16];

  const int lane = threadIdx.x;
  const int qt = blockIdx.x;
  const int h  = blockIdx.y;
  const int b  = blockIdx.z;
  const int half = lane >> 4;
  const int ln   = lane & 15;
  const int kb   = half * 8;

  const __bf16* qrow =
      Q + ((size_t)(b * LT + qt * 16 + ln)) * HIDDEN + h * HEAD_DIM;
  v16bf qf[8];
#pragma unroll
  for (int ks = 0; ks < 8; ++ks) qf[ks] = ldfrag(qrow + ks * 32 + kb);

  for (int kt = 0; kt < LI / 16; ++kt) {
    const __bf16* krow =
        K + ((size_t)(b * LI + kt * 16 + ln)) * HIDDEN + h * HEAD_DIM;
    v8f acc = v8f{};
#pragma unroll
    for (int ks = 0; ks < 8; ++ks) {
      v16bf bf = ldfrag(krow + ks * 32 + kb);
      acc = __builtin_amdgcn_wmma_f32_16x16x32_bf16(
          false, qf[ks], false, bf, (short)0, acc, false, false);
    }
#pragma unroll
    for (int v = 0; v < 8; ++v) {
      int m = v + 8 * half;
      S[m * LI + kt * 16 + ln] = acc[v] * 0.0625f;  // 1/sqrt(256)
    }
  }
  __syncthreads();

  for (int r = 0; r < 16; ++r) {
    float mx = -3.0e38f;
    for (int i = lane; i < LI; i += 32) mx = fmaxf(mx, S[r * LI + i]);
#pragma unroll
    for (int off = 16; off; off >>= 1) mx = fmaxf(mx, __shfl_xor(mx, off, 32));
    float sum = 0.0f;
    for (int i = lane; i < LI; i += 32) {
      float e = __expf(S[r * LI + i] - mx);
      P[r * LI + i] = f2bf(e);
      sum += e;
    }
#pragma unroll
    for (int off = 16; off; off >>= 1) sum += __shfl_xor(sum, off, 32);
    if (lane == 0) rinv[r] = 1.0f / sum;
  }
  __syncthreads();

  for (int dt = 0; dt < HEAD_DIM / 16; ++dt) {
    const __bf16* vrow =
        Vt + ((size_t)((b * HEADS + h) * HEAD_DIM + dt * 16 + ln)) * LI;
    v8f acc = v8f{};
    for (int k2 = 0; k2 < LI / 32; ++k2) {
      v16bf af = ldfrag(&P[ln * LI + k2 * 32 + kb]);
      v16bf bf = ldfrag(vrow + k2 * 32 + kb);
      acc = __builtin_amdgcn_wmma_f32_16x16x32_bf16(
          false, af, false, bf, (short)0, acc, false, false);
    }
#pragma unroll
    for (int v = 0; v < 8; ++v) {
      int m = v + 8 * half;
      float o = acc[v] * rinv[m];
      X[((size_t)(b * LT + qt * 16 + m)) * HIDDEN + h * HEAD_DIM + dt * 16 + ln] =
          f2bf(o);
    }
  }
}

// ---------------------------------------------------------------------------
// LayerNorm over last dim (768), eps=1e-5. One block (256 threads) per row.
// ---------------------------------------------------------------------------
__global__ __launch_bounds__(256) void layernorm_kernel(
    const float* __restrict__ Xin, const float* __restrict__ gamma,
    const float* __restrict__ beta, float* __restrict__ Out) {
  __shared__ float sw[8], ssw[8];
  __shared__ float mu_s, rstd_s;
  const int row = blockIdx.x;
  const int t = threadIdx.x;
  float vals[3];
  float s = 0.0f, ss = 0.0f;
#pragma unroll
  for (int i = 0; i < 3; ++i) {
    float v = Xin[(size_t)row * TEXT_DIM + t + i * 256];
    vals[i] = v;
    s += v;
    ss += v * v;
  }
#pragma unroll
  for (int off = 16; off; off >>= 1) {
    s += __shfl_xor(s, off, 32);
    ss += __shfl_xor(ss, off, 32);
  }
  if ((t & 31) == 0) { sw[t >> 5] = s; ssw[t >> 5] = ss; }
  __syncthreads();
  if (t == 0) {
    float S = 0.0f, SS = 0.0f;
#pragma unroll
    for (int i = 0; i < 8; ++i) { S += sw[i]; SS += ssw[i]; }
    float mu = S * (1.0f / TEXT_DIM);
    float var = SS * (1.0f / TEXT_DIM) - mu * mu;
    mu_s = mu;
    rstd_s = rsqrtf(var + 1e-5f);
  }
  __syncthreads();
#pragma unroll
  for (int i = 0; i < 3; ++i) {
    int c = t + i * 256;
    Out[(size_t)row * TEXT_DIM + c] = (vals[i] - mu_s) * rstd_s * gamma[c] + beta[c];
  }
}

// ---------------------------------------------------------------------------
// Host-side launch
// ---------------------------------------------------------------------------
extern "C" void kernel_launch(void* const* d_in, const int* in_sizes, int n_in,
                              void* d_out, int out_size, void* d_ws, size_t ws_size,
                              hipStream_t stream) {
  (void)in_sizes; (void)n_in; (void)out_size; (void)ws_size;
  const float* text  = (const float*)d_in[0];
  const float* image = (const float*)d_in[1];
  const float* wq = (const float*)d_in[2];  const float* bq = (const float*)d_in[3];
  const float* wk = (const float*)d_in[4];  const float* bk = (const float*)d_in[5];
  const float* wv = (const float*)d_in[6];  const float* bv = (const float*)d_in[7];
  const float* wr = (const float*)d_in[8];  const float* br = (const float*)d_in[9];
  const float* w1 = (const float*)d_in[10]; const float* b1 = (const float*)d_in[11];
  const float* w2 = (const float*)d_in[12]; const float* b2 = (const float*)d_in[13];
  const float* gamma = (const float*)d_in[14];
  const float* beta  = (const float*)d_in[15];
  float* out = (float*)d_out;

  const int MT = BATCH * LT;   // 8192 text rows
  const int MI = BATCH * LI;   // 9216 image rows

  char* cur = (char*)d_ws;
  auto alloc = [&](size_t bytes) {
    void* p = cur;
    cur += (bytes + 255) & ~(size_t)255;
    return p;
  };
  __bf16* text_bf  = (__bf16*)alloc((size_t)MT * TEXT_DIM * 2);
  __bf16* image_bf = (__bf16*)alloc((size_t)MI * IMAGE_DIM * 2);
  __bf16* wq_t = (__bf16*)alloc((size_t)HIDDEN * TEXT_DIM * 2);
  __bf16* wk_t = (__bf16*)alloc((size_t)HIDDEN * IMAGE_DIM * 2);
  __bf16* wv_t = (__bf16*)alloc((size_t)HIDDEN * IMAGE_DIM * 2);
  __bf16* wr_t = (__bf16*)alloc((size_t)TEXT_DIM * HIDDEN * 2);
  __bf16* w1_t = (__bf16*)alloc((size_t)FFD * TEXT_DIM * 2);
  __bf16* w2_t = (__bf16*)alloc((size_t)TEXT_DIM * FFD * 2);
  __bf16* Qb   = (__bf16*)alloc((size_t)MT * HIDDEN * 2);
  __bf16* Kb   = (__bf16*)alloc((size_t)MI * HIDDEN * 2);
  __bf16* Vt   = (__bf16*)alloc((size_t)MI * HIDDEN * 2);
  __bf16* Xb   = (__bf16*)alloc((size_t)MT * HIDDEN * 2);
  __bf16* OutB = (__bf16*)alloc((size_t)MT * TEXT_DIM * 2);
  __bf16* FF1  = (__bf16*)alloc((size_t)MT * FFD * 2);
  float*  OutF  = (float*)alloc((size_t)MT * TEXT_DIM * 4);
  float*  PreLN = (float*)alloc((size_t)MT * TEXT_DIM * 4);

  // 1) activation converts
  {
    int n = MT * TEXT_DIM;
    cvt_f32_bf16<<<(n + 255) / 256, 256, 0, stream>>>(text, text_bf, n);
    n = MI * IMAGE_DIM;
    cvt_f32_bf16<<<(n + 255) / 256, 256, 0, stream>>>(image, image_bf, n);
  }
  // 2) weight transposes -> [N,K] bf16
  {
    int n;
    n = TEXT_DIM * HIDDEN;
    transpose_cvt<<<(n + 255) / 256, 256, 0, stream>>>(wq, wq_t, TEXT_DIM, HIDDEN);
    n = IMAGE_DIM * HIDDEN;
    transpose_cvt<<<(n + 255) / 256, 256, 0, stream>>>(wk, wk_t, IMAGE_DIM, HIDDEN);
    transpose_cvt<<<(n + 255) / 256, 256, 0, stream>>>(wv, wv_t, IMAGE_DIM, HIDDEN);
    n = HIDDEN * TEXT_DIM;
    transpose_cvt<<<(n + 255) / 256, 256, 0, stream>>>(wr, wr_t, HIDDEN, TEXT_DIM);
    n = TEXT_DIM * FFD;
    transpose_cvt<<<(n + 255) / 256, 256, 0, stream>>>(w1, w1_t, TEXT_DIM, FFD);
    transpose_cvt<<<(n + 255) / 256, 256, 0, stream>>>(w2, w2_t, FFD, TEXT_DIM);
  }
  // 3) projections
  gemm_wmma_tdm<<<dim3(HIDDEN / 64, MT / 64), 128, 0, stream>>>(
      text_bf, wq_t, bq, nullptr, Qb, nullptr, MT, HIDDEN, TEXT_DIM, 0, 0);
  gemm_wmma_tdm<<<dim3(HIDDEN / 64, MI / 64), 128, 0, stream>>>(
      image_bf, wk_t, bk, nullptr, Kb, nullptr, MI, HIDDEN, IMAGE_DIM, 0, 0);
  gemm_wmma_tdm<<<dim3(HIDDEN / 64, MI / 64), 128, 0, stream>>>(
      image_bf, wv_t, bv, nullptr, Vt, nullptr, MI, HIDDEN, IMAGE_DIM, 0, 1);
  // 4) attention
  attn_kernel<<<dim3(LT / 16, HEADS, BATCH), 32, 0, stream>>>(Qb, Kb, Vt, Xb);
  // 5) output projection (fp32 for residual/LN + bf16 for FF input)
  gemm_wmma_tdm<<<dim3(TEXT_DIM / 64, MT / 64), 128, 0, stream>>>(
      Xb, wr_t, br, OutF, OutB, nullptr, MT, TEXT_DIM, HIDDEN, 0, 0);
  // 6) FF
  gemm_wmma_tdm<<<dim3(FFD / 64, MT / 64), 128, 0, stream>>>(
      OutB, w1_t, b1, nullptr, FF1, nullptr, MT, FFD, TEXT_DIM, 1, 0);
  gemm_wmma_tdm<<<dim3(TEXT_DIM / 64, MT / 64), 128, 0, stream>>>(
      FF1, w2_t, b2, PreLN, nullptr, OutF, MT, TEXT_DIM, FFD, 0, 0);
  // 7) LayerNorm
  layernorm_kernel<<<MT, 256, 0, stream>>>(PreLN, gamma, beta, out);
}
